// MLP_Head_79319456022592
// MI455X (gfx1250) — matrix operand order repaired and checked by reading, not compiled, not run on hardware
//
#include <hip/hip_runtime.h>
#include <stdint.h>

// Problem sizes (fixed by the reference)
#define T_STEPS 128
#define BATCH   512
#define DIM     512
#define OUT     10

typedef __attribute__((ext_vector_type(16))) __bf16 v16bf;
typedef __attribute__((ext_vector_type(8)))  __bf16 v8bf;
typedef __attribute__((ext_vector_type(8)))  float  v8f;

// ---------- helpers ----------
__device__ __forceinline__ unsigned short f2bf(float f) {
    unsigned u = __float_as_uint(f);
    u += 0x7FFFu + ((u >> 16) & 1u);   // round-to-nearest-even
    return (unsigned short)(u >> 16);
}

// CDNA5 async copy: global -> LDS, 16B per lane, tracked by ASYNCcnt.
// LDS address operand = low 32 bits of the generic shared pointer
// (flat->LDS mapping truncates to addr[31:0]).
__device__ __forceinline__ void async_load_b128(unsigned short* lds_ptr,
                                                const unsigned short* gptr) {
    unsigned l = (unsigned)(unsigned long long)lds_ptr;
    asm volatile("global_load_async_to_lds_b128 %0, %1, off"
                 :: "v"(l), "v"((unsigned long long)gptr)
                 : "memory");
}
#define WAIT_ASYNC(n) asm volatile("s_wait_asynccnt " #n ::: "memory")

// ---------- kernel 1: X = bf16(mask * data), stored time-major [T*B, D] ----------
__global__ __launch_bounds__(256)
void pack_x(const float* __restrict__ data, const float* __restrict__ mask,
            unsigned short* __restrict__ Xb) {
    int i = blockIdx.x * 256 + threadIdx.x;     // one float4 chunk per thread
    int c = i & 127;                            // D/4 = 128 chunks per row
    int b = (i >> 7) & 511;
    int t = i >> 16;                            // / (128*512)
    const float4 d4 = *(const float4*)(&data[((size_t)b * T_STEPS + t) * DIM + c * 4]);
    const float4 m4 = *(const float4*)(&mask[((size_t)t * BATCH + b) * DIM + c * 4]);
    ushort4 r;
    r.x = f2bf(d4.x * m4.x); r.y = f2bf(d4.y * m4.y);
    r.z = f2bf(d4.z * m4.z); r.w = f2bf(d4.w * m4.w);
    *(ushort4*)(&Xb[((size_t)t * BATCH + b) * DIM + c * 4]) = r;
}

// ---------- kernel 2: W1 -> bf16 ----------
__global__ __launch_bounds__(256)
void pack_w(const float* __restrict__ src, unsigned short* __restrict__ dst) {
    int i = blockIdx.x * 256 + threadIdx.x;
    const float4 v = *(const float4*)(&src[(size_t)i * 4]);
    ushort4 r;
    r.x = f2bf(v.x); r.y = f2bf(v.y); r.z = f2bf(v.z); r.w = f2bf(v.w);
    *(ushort4*)(&dst[(size_t)i * 4]) = r;
}

// ---------- kernel 3: Y[65536,512] = Xb @ W1b^T, f32 accumulate via WMMA ----------
// Block tile 128x128, K-step 32, 8 waves as 4x2 -> 32x64 per wave (2x4 WMMA tiles).
// Double-buffered LDS staged by GLOBAL_LOAD_ASYNC_TO_LDS_B128.
#define M_BLK 128
#define N_BLK 128
#define K_BLK 32
#define LDT 40     // 32 + 8 pad (ushorts): 80B row stride, bank-conflict-free, 16B aligned

__device__ __forceinline__ v16bf load_frag_a(const unsigned short* T, int row, int half) {
    // ISA 16-bit A layout (16x32): lanes 0-15 hold K{0..7,16..23}; lanes 16-31 K{8..15,24..31}
    const unsigned short* p = T + row * LDT + half * 8;
    v8bf lo = *(const v8bf*)(p);
    v8bf hi = *(const v8bf*)(p + 16);
    return __builtin_shufflevector(lo, hi, 0,1,2,3,4,5,6,7,8,9,10,11,12,13,14,15);
}
__device__ __forceinline__ v16bf load_frag_b(const unsigned short* T, int row, int half) {
    // 16-bit B layout (32x16): lanes 0-15 = cols, K{0..7,16..23}; lanes 16-31 K{8..15,24..31}
    const unsigned short* p = T + row * LDT + half * 8;
    v8bf lo = *(const v8bf*)(p);
    v8bf hi = *(const v8bf*)(p + 16);
    return __builtin_shufflevector(lo, hi, 0,1,2,3,4,5,6,7,8,9,10,11,12,13,14,15);
}
__device__ __forceinline__ void store_tile(float* __restrict__ Y, int rowBase, int col,
                                           int half, v8f c) {
    #pragma unroll
    for (int r = 0; r < 8; ++r)
        Y[(size_t)(rowBase + half * 8 + r) * DIM + col] = c[r];
}

__global__ __launch_bounds__(256)
void gemm_bf16_wmma(const unsigned short* __restrict__ Xb,
                    const unsigned short* __restrict__ W1b,
                    float* __restrict__ Y) {
    __shared__ __align__(16) unsigned short As[2][M_BLK * LDT];
    __shared__ __align__(16) unsigned short Bs[2][N_BLK * LDT];

    const int tid  = threadIdx.x;
    const int lane = tid & 31;
    const int wid  = tid >> 5;     // 8 waves
    const int wm   = wid >> 1;     // 0..3 -> 32-row stripe
    const int wn   = wid & 1;      // 0..1 -> 64-col stripe
    const int half = lane >> 4;
    const int l16  = lane & 15;

    const int mBase = blockIdx.y * M_BLK;
    const int nBase = blockIdx.x * N_BLK;

    // Per-thread staging assignment: 2 A chunks + 2 B chunks (16B each) per K-step.
    const int rowA0 = tid >> 2,            kcA0 = (tid & 3) * 8;          // chunks 0..255
    const int rowA1 = (tid + 256) >> 2,    kcA1 = kcA0;                   // chunks 256..511
    // B uses the same decomposition (128 rows x 4 chunks).
    const unsigned short* gA0 = Xb  + (size_t)(mBase + rowA0) * DIM + kcA0;
    const unsigned short* gA1 = Xb  + (size_t)(mBase + rowA1) * DIM + kcA1;
    const unsigned short* gB0 = W1b + (size_t)(nBase + rowA0) * DIM + kcA0;
    const unsigned short* gB1 = W1b + (size_t)(nBase + rowA1) * DIM + kcA1;

    v8f acc[2][4];
    #pragma unroll
    for (int i = 0; i < 2; ++i)
        #pragma unroll
        for (int j = 0; j < 4; ++j) acc[i][j] = {};

    // Prologue: stage k0 = 0 into buffer 0 (4 async loads per thread).
    async_load_b128(&As[0][rowA0 * LDT + kcA0], gA0);
    async_load_b128(&Bs[0][rowA0 * LDT + kcA0], gB0);
    async_load_b128(&As[0][rowA1 * LDT + kcA1], gA1);
    async_load_b128(&Bs[0][rowA1 * LDT + kcA1], gB1);

    int buf = 0;
    for (int k0 = 0; k0 < DIM; k0 += K_BLK) {
        if (k0 + K_BLK < DIM) {
            // Stage next K-slab into the other buffer while we compute this one.
            const int nb = buf ^ 1, ko = k0 + K_BLK;
            async_load_b128(&As[nb][rowA0 * LDT + kcA0], gA0 + ko);
            async_load_b128(&Bs[nb][rowA0 * LDT + kcA0], gB0 + ko);
            async_load_b128(&As[nb][rowA1 * LDT + kcA1], gA1 + ko);
            async_load_b128(&Bs[nb][rowA1 * LDT + kcA1], gB1 + ko);
            WAIT_ASYNC(4);   // previous stage (4 loads, in-order) has landed
        } else {
            WAIT_ASYNC(0);
        }
        __syncthreads();     // all waves' staging for `buf` complete

        v16bf a0 = load_frag_a(As[buf], wm * 32 + l16,      half);
        v16bf a1 = load_frag_a(As[buf], wm * 32 + 16 + l16, half);
        v16bf b0 = load_frag_b(Bs[buf], wn * 64 + l16,      half);
        v16bf b1 = load_frag_b(Bs[buf], wn * 64 + 16 + l16, half);
        v16bf b2 = load_frag_b(Bs[buf], wn * 64 + 32 + l16, half);
        v16bf b3 = load_frag_b(Bs[buf], wn * 64 + 48 + l16, half);

        acc[0][0] = __builtin_amdgcn_wmma_f32_16x16x32_bf16(false, a0, false, b0, (short)0, acc[0][0], false, false);
        acc[0][1] = __builtin_amdgcn_wmma_f32_16x16x32_bf16(false, a0, false, b1, (short)0, acc[0][1], false, false);
        acc[0][2] = __builtin_amdgcn_wmma_f32_16x16x32_bf16(false, a0, false, b2, (short)0, acc[0][2], false, false);
        acc[0][3] = __builtin_amdgcn_wmma_f32_16x16x32_bf16(false, a0, false, b3, (short)0, acc[0][3], false, false);
        acc[1][0] = __builtin_amdgcn_wmma_f32_16x16x32_bf16(false, a1, false, b0, (short)0, acc[1][0], false, false);
        acc[1][1] = __builtin_amdgcn_wmma_f32_16x16x32_bf16(false, a1, false, b1, (short)0, acc[1][1], false, false);
        acc[1][2] = __builtin_amdgcn_wmma_f32_16x16x32_bf16(false, a1, false, b2, (short)0, acc[1][2], false, false);
        acc[1][3] = __builtin_amdgcn_wmma_f32_16x16x32_bf16(false, a1, false, b3, (short)0, acc[1][3], false, false);

        __syncthreads();     // done reading `buf` before it is re-staged next+1 iter
        buf ^= 1;
    }

    const int r0 = mBase + wm * 32;
    const int c0 = nBase + wn * 64 + l16;
    #pragma unroll
    for (int tm = 0; tm < 2; ++tm)
        #pragma unroll
        for (int tn = 0; tn < 4; ++tn)
            store_tile(Y, r0 + tm * 16, c0 + tn * 16, half, acc[tm][tn]);
}

// ---------- kernel 4: BN stats folded into per-(t,d) scale/shift ----------
// b1 cancels exactly under training-mode BatchNorm (mean subtraction) -> ignored.
__global__ __launch_bounds__(256)
void bn_stats(const float* __restrict__ Y,
              const float* __restrict__ gamma, const float* __restrict__ bn_beta,
              float* __restrict__ scale, float* __restrict__ shift) {
    int idx = blockIdx.x * 256 + threadIdx.x;  // t*DIM + d
    int d   = idx & (DIM - 1);
    const float* base = Y + (size_t)(idx >> 9) * BATCH * DIM + d;  // t = idx/512
    float s = 0.f, sq = 0.f;
    for (int b = 0; b < BATCH; ++b) {          // coalesced across the wave (d contiguous)
        float v = base[(size_t)b * DIM];
        s += v; sq += v * v;
    }
    float mean = s * (1.0f / BATCH);
    float var  = sq * (1.0f / BATCH) - mean * mean;
    float sc   = gamma[d] * rsqrtf(var + 1e-5f);
    scale[idx] = sc;
    shift[idx] = bn_beta[d] - mean * sc;
}

// ---------- kernel 5: sequential LIF scan over T; one block per batch element ----------
__global__ __launch_bounds__(512)
void lif_scan(const float* __restrict__ Y,
              const float* __restrict__ scale, const float* __restrict__ shift,
              const float* __restrict__ W2, const float* __restrict__ b2,
              float* __restrict__ out) {
    __shared__ float    W2s[OUT * DIM];   // 20 KB
    __shared__ unsigned masks[16];
    __shared__ float    red[OUT * 16];

    const int tid  = threadIdx.x;        // = d (0..511)
    const int b    = blockIdx.x;
    const int lane = tid & 31;
    const int w    = tid >> 5;

    for (int i = tid; i < OUT * DIM; i += 512) W2s[i] = W2[i];
    __syncthreads();

    float mem1 = 0.f, spk1 = 0.f;
    float mem2 = 0.f, spk2 = 0.f;        // live only on tid < OUT

    for (int t = 0; t < T_STEPS; ++t) {
        float cur1 = Y[((size_t)t * BATCH + b) * DIM + tid] * scale[t * DIM + tid]
                   + shift[t * DIM + tid];
        if (t + 1 < T_STEPS)
            __builtin_prefetch(&Y[((size_t)(t + 1) * BATCH + b) * DIM + tid], 0, 0);

        mem1 = 0.5f * mem1 + cur1 - spk1;          // Leaky + subtract reset (THR=1)
        spk1 = (mem1 > 1.0f) ? 1.0f : 0.0f;

        // wave32 ballot of binary spikes via shfl-xor OR-reduce
        unsigned bit = (spk1 != 0.f) ? (1u << lane) : 0u;
        #pragma unroll
        for (int off = 16; off > 0; off >>= 1)
            bit |= (unsigned)__shfl_xor((int)bit, off, 32);
        if (lane == 0) masks[w] = bit;
        __syncthreads();

        // layer 2: cur2[o] = sum over set spike bits of W2[o, d]
        if (tid < OUT * 16) {
            int o = tid >> 4, ww = tid & 15;
            unsigned m = masks[ww];
            float s = 0.f;
            while (m) { int j = __builtin_ctz(m); m &= m - 1u; s += W2s[o * DIM + ww * 32 + j]; }
            red[o * 16 + ww] = s;
        }
        __syncthreads();

        if (tid < OUT) {
            float c2 = b2[tid];
            #pragma unroll
            for (int ww = 0; ww < 16; ++ww) c2 += red[tid * 16 + ww];
            mem2 = 0.5f * mem2 + c2 - spk2;
            spk2 = (mem2 > 1.0f) ? 1.0f : 0.0f;
            out[((size_t)t * BATCH + b) * OUT + tid] = spk2;
        }
    }
}

// ---------- launcher ----------
extern "C" void kernel_launch(void* const* d_in, const int* in_sizes, int n_in,
                              void* d_out, int out_size, void* d_ws, size_t ws_size,
                              hipStream_t stream) {
    const float* data    = (const float*)d_in[0];
    const float* mask    = (const float*)d_in[1];
    const float* W1      = (const float*)d_in[2];
    // d_in[3] = b1: cancels under training-mode BatchNorm -> unused
    const float* gamma   = (const float*)d_in[4];
    const float* bn_beta = (const float*)d_in[5];
    const float* W2      = (const float*)d_in[6];
    const float* b2      = (const float*)d_in[7];
    float* out = (float*)d_out;

    // Workspace layout (~193.5 MB)
    char* ws = (char*)d_ws;
    unsigned short* Xb  = (unsigned short*)(ws);                               // 64 MB
    unsigned short* W1b = (unsigned short*)(ws + ((size_t)64 << 20));          // 512 KB (pad to 1 MB)
    float* Y     = (float*)(ws + ((size_t)65 << 20));                          // 128 MB
    float* scale = (float*)(ws + ((size_t)65 << 20) + ((size_t)128 << 20));    // 256 KB
    float* shift = scale + T_STEPS * DIM;                                      // 256 KB

    pack_x<<<(T_STEPS * BATCH * (DIM / 4)) / 256, 256, 0, stream>>>(data, mask, Xb);
    pack_w<<<(DIM * DIM / 4) / 256, 256, 0, stream>>>(W1, W1b);

    dim3 g(DIM / N_BLK, (T_STEPS * BATCH) / M_BLK);   // (4, 512)
    gemm_bf16_wmma<<<g, 256, 0, stream>>>(Xb, W1b, Y);

    bn_stats<<<(T_STEPS * DIM) / 256, 256, 0, stream>>>(Y, gamma, bn_beta, scale, shift);
    lif_scan<<<BATCH, 512, 0, stream>>>(Y, scale, shift, W2, b2, out);
}
